// BiLSTM_CRF_44933947851270
// MI455X (gfx1250) — compile-verified, matrix operand order checked by reading, not run on hardware
//
#include <hip/hip_runtime.h>
#include <hip/hip_bf16.h>

// ---------------------------------------------------------------------------
// BiLSTM-CRF forward for MI455X (gfx1250, wave32, WMMA f16 -> f32 accum,
// TDM tensor_load_to_lds for GEMM A-tile staging)
// ---------------------------------------------------------------------------
#define Bb   64
#define Tt   512
#define Ee   256
#define HIDh 512
#define Hh   256
#define Kk   32
#define G4H  1024      // 4*H gate width
#define NEGV (-10000.0f)

typedef _Float16 half_t;
typedef __attribute__((ext_vector_type(16))) _Float16 v16h;
typedef __attribute__((ext_vector_type(8)))  _Float16 v8h;
typedef __attribute__((ext_vector_type(8)))  float    v8f;
typedef __attribute__((ext_vector_type(4)))  unsigned int u32x4;
typedef __attribute__((ext_vector_type(8)))  int      i32x8;
typedef __attribute__((ext_vector_type(4)))  int      i32x4;

#if defined(__has_builtin)
#if __has_builtin(__builtin_amdgcn_tensor_load_to_lds)
#define HAVE_TDM 1
#endif
#endif

__device__ __forceinline__ float sigf(float x) { return 1.f / (1.f + expf(-x)); }

// D = A(16x32 f16) * B(32x16 f16) + C(16x16 f32), wave32 WMMA
__device__ __forceinline__ v8f wmma_f16(v16h a, v16h b, v8f c) {
    return __builtin_amdgcn_wmma_f32_16x16x32_f16(
        /*neg_a=*/false, a, /*neg_b=*/false, b,
        /*c_mod=*/(short)0, c, /*reuse_a=*/false, /*reuse_b=*/false);
}

// build a v16h fragment from two 16-byte runs
__device__ __forceinline__ v16h ld_frag2(const half_t* p0, const half_t* p1) {
    v8h lo = *(const v8h*)p0;
    v8h hi = *(const v8h*)p1;
    v16h r;
#pragma unroll
    for (int i = 0; i < 8; ++i) { r[i] = lo[i]; r[i + 8] = hi[i]; }
    return r;
}

#ifdef HAVE_TDM
// TDM: DMA a 2D tile (tileW elems of elemSize=2B per row, tileH rows,
// row stride strideElems) from global memory into LDS. One issue per caller.
__device__ __forceinline__ void tdm_load_2d_f16(unsigned lds_addr,
                                                const half_t* gptr,
                                                int tileW, int tileH,
                                                int strideElems) {
    unsigned long long ga = (unsigned long long)(uintptr_t)gptr;
    u32x4 g0;
    g0[0] = 1u;                                  // count=1, user descriptor
    g0[1] = lds_addr;                            // lds_addr [63:32]
    g0[2] = (unsigned)(ga & 0xffffffffu);        // global_addr [95:64]
    g0[3] = (unsigned)((ga >> 32) & 0x1ffffffu)  // global_addr [120:96]
            | (2u << 30);                        // type=2 ("image")
    i32x8 g1;
    g1[0] = (1 << 16);                           // wg_mask=0, data_size=1 (2B)
    g1[1] = (int)((unsigned)(tileW & 0xffff) << 16);      // tensor_dim0 lo16
    g1[2] = (int)(((unsigned)tileH & 0xffffu) << 16);     // dim0 hi=0 | tensor_dim1 lo16
    g1[3] = (int)((unsigned)(tileW & 0xffff) << 16);      // tensor_dim1 hi=0 | tile_dim0
    g1[4] = tileH & 0xffff;                      // tile_dim1 | tile_dim2=0
    g1[5] = strideElems;                         // tensor_dim0_stride lo32
    g1[6] = 0;                                   // stride0 hi | stride1 lo
    g1[7] = 0;                                   // stride1 hi
    i32x4 gz = {0, 0, 0, 0};
#if __clang_major__ >= 23
    i32x8 gz8 = {0, 0, 0, 0, 0, 0, 0, 0};
    __builtin_amdgcn_tensor_load_to_lds(g0, g1, gz, gz, gz8, 0);
#else
    __builtin_amdgcn_tensor_load_to_lds(g0, g1, gz, gz, 0);
#endif
}
#endif

// ---------------------------------------------------------------------------
// Pack f32 weight [N,K] row-major into per-WMMA-fragment B layout:
// frag(nt,kt) occupies 512 f16; lane owns 16 contiguous f16 (coalesced loads).
// B frag mapping: lane -> n = nt*16+(lane&15); elem e -> k = kt*32 + hi*8 +
// (e&7) + (e>=8 ? 16 : 0), hi = lane>>4   (per ISA 16-bit 32x16 layout)
// ---------------------------------------------------------------------------
__global__ void k_pack_w(const float* __restrict__ src, half_t* __restrict__ dst,
                         int N, int Kd) {
    int kTiles = Kd >> 5;
    int total = N * Kd;
    for (int idx = blockIdx.x * blockDim.x + threadIdx.x; idx < total;
         idx += gridDim.x * blockDim.x) {
        int frag = idx >> 9;
        int r    = idx & 511;
        int lane = r >> 4;
        int e    = r & 15;
        int nt   = frag / kTiles;
        int kt   = frag - nt * kTiles;
        int n    = (nt << 4) + (lane & 15);
        int hi   = lane >> 4;
        int k    = (kt << 5) + (hi << 3) + (e & 7) + ((e & 8) ? 16 : 0);
        dst[idx] = (half_t)src[(size_t)n * Kd + k];
    }
}

// x0[(t*B+b), e] = (f16) emb[sentence[b,t], e]
__global__ void k_embed(const int* __restrict__ sent, const float* __restrict__ emb,
                        half_t* __restrict__ x0) {
    int r = blockIdx.x;            // t*B + b
    int t = r >> 6, b = r & 63;
    int tok = sent[b * Tt + t];
    int e = threadIdx.x;           // 256 threads == Ee
    x0[(size_t)r * Ee + e] = (half_t)emb[(size_t)tok * Ee + e];
}

__global__ void k_mask(const int* __restrict__ seq_len, float* __restrict__ mask) {
    int idx = blockIdx.x * blockDim.x + threadIdx.x;   // t*64 + b
    if (idx < Tt * Bb) {
        int b = idx & 63, t = idx >> 6;
        mask[idx] = (t < seq_len[b]) ? 1.f : 0.f;
    }
}

// ---------------------------------------------------------------------------
// XG[m,n] = sum_k A[m,k]*W[n,k] + bias[n]   (A f16 [M=32768,K], W packed)
// block = 256 thr (8 waves). Block tile 128(M) x 128(N); wave = one M-tile,
// 8 N-tiles. A chunk staged in LDS via TDM tensor_load_to_lds (TENSORcnt);
// B frags streamed coalesced from L2.
// ---------------------------------------------------------------------------
__global__ __launch_bounds__(256) void k_gemm(const half_t* __restrict__ A,
                                              const half_t* __restrict__ Wp,
                                              const float* __restrict__ bias,
                                              float* __restrict__ out, int Kd) {
    __shared__ half_t sA[128 * 32];
    int kTiles = Kd >> 5;
    int tid  = threadIdx.x;
    int wave = tid >> 5, lane = tid & 31;
    int ln = lane & 15, hi = lane >> 4;
    int rowBase = blockIdx.x * 128;
    int ntBase  = blockIdx.y * 8;

    v8f acc[8];
#pragma unroll
    for (int i = 0; i < 8; ++i) {
        float bv = bias[(ntBase + i) * 16 + ln];
#pragma unroll
        for (int v = 0; v < 8; ++v) acc[i][v] = bv;
    }

    for (int kc = 0; kc < kTiles; ++kc) {
        __syncthreads();               // previous iteration's sA reads done
#ifdef HAVE_TDM
        if (wave == 0) {
            // DMA the 128x32 f16 A-tile into LDS with the Tensor Data Mover
            tdm_load_2d_f16((unsigned)(uintptr_t)&sA[0],
                            A + (size_t)rowBase * Kd + (kc << 5),
                            /*tileW=*/32, /*tileH=*/128, /*stride=*/Kd);
            __builtin_amdgcn_s_wait_tensorcnt(0);
        }
#else
        {   // fallback: stage A tile 128x32 f16 with vector loads
            int rrow = tid >> 1;
            int hf   = tid & 1;
            const uint4* src =
                (const uint4*)(A + (size_t)(rowBase + rrow) * Kd + (kc << 5) + hf * 16);
            *(uint4*)(sA + rrow * 32 + hf * 16) = *src;
        }
#endif
        __syncthreads();               // tile visible to all waves
        // A fragment for this wave's M-tile
        const half_t* ap = sA + (wave * 16 + ln) * 32 + (hi << 3);
        v16h af = ld_frag2(ap, ap + 16);
#pragma unroll
        for (int i = 0; i < 8; ++i) {
            int nt = ntBase + i;
            const half_t* bp = Wp + ((size_t)(nt * kTiles + kc) << 9) + (lane << 4);
            if (kc + 1 < kTiles)
                __builtin_prefetch(Wp + ((size_t)(nt * kTiles + kc + 1) << 9) + (lane << 4), 0, 1);
            v16h bf = ld_frag2(bp, bp + 8);
            acc[i] = wmma_f16(af, bf, acc[i]);
        }
    }
    // D layout: lane -> n, vgpr v -> m = hi*8 + v
#pragma unroll
    for (int i = 0; i < 8; ++i) {
        int n = (ntBase + i) * 16 + ln;
#pragma unroll
        for (int v = 0; v < 8; ++v) {
            int m = rowBase + wave * 16 + (hi << 3) + v;
            out[(size_t)m * G4H + n] = acc[i][v];
        }
    }
}

// ---------------------------------------------------------------------------
// Sequential LSTM scan, one persistent 1024-thread workgroup per direction.
// Per step: G = xg[t] + h @ Whh^T via WMMA, then gate math on accumulators.
// Wave w: jt = w&15 (hidden cols 16*jt..), mhalf = w>>4 (batch halves);
// wave owns all four gate tiles for its columns -> no gate staging in LDS.
// ---------------------------------------------------------------------------
__global__ __launch_bounds__(1024) void k_scan(const float* __restrict__ xgF,
                                               const float* __restrict__ xgB,
                                               const half_t* __restrict__ WF,
                                               const half_t* __restrict__ WB,
                                               const float* __restrict__ mask,
                                               half_t* __restrict__ y) {
    __shared__ half_t sH[64 * 256];     // h state, f16, [b][j]
    int dir = blockIdx.x;
    const float*  xg = dir ? xgB : xgF;
    const half_t* Wp = dir ? WB : WF;
    int colOff = dir ? Hh : 0;

    int tid  = threadIdx.x;
    int wave = tid >> 5, lane = tid & 31;
    int jt = wave & 15, mhalf = wave >> 4;
    int ln = lane & 15, hi = lane >> 4;

    for (int i = tid; i < 64 * 256; i += 1024) sH[i] = (half_t)0.f;
    float cReg[2][8];
#pragma unroll
    for (int mi = 0; mi < 2; ++mi)
#pragma unroll
        for (int v = 0; v < 8; ++v) cReg[mi][v] = 0.f;
    __syncthreads();

    for (int step = 0; step < Tt; ++step) {
        int t = dir ? (Tt - 1 - step) : step;
        v8f acc[2][4];
        // init accumulators from precomputed input-gate preactivations
#pragma unroll
        for (int mi = 0; mi < 2; ++mi) {
            int mt = mhalf * 2 + mi;
#pragma unroll
            for (int g = 0; g < 4; ++g) {
                int n = (g * 16 + jt) * 16 + ln;        // gate col in [0,1024)
#pragma unroll
                for (int v = 0; v < 8; ++v) {
                    int m = mt * 16 + hi * 8 + v;       // batch row
                    acc[mi][g][v] = xg[(size_t)(t * 64 + m) * G4H + n];
                }
            }
        }
        // recurrent GEMM: K = 256 in 8 WMMA chunks
#pragma unroll 1
        for (int kt = 0; kt < 8; ++kt) {
            v16h af[2];
#pragma unroll
            for (int mi = 0; mi < 2; ++mi) {
                int mt = mhalf * 2 + mi;
                const half_t* ap = sH + (mt * 16 + ln) * 256 + kt * 32 + hi * 8;
                af[mi] = ld_frag2(ap, ap + 16);
            }
#pragma unroll
            for (int g = 0; g < 4; ++g) {
                int nt = g * 16 + jt;
                const half_t* bp = Wp + ((size_t)(nt * 8 + kt) << 9) + (lane << 4);
                v16h bf = ld_frag2(bp, bp + 8);
                acc[0][g] = wmma_f16(af[0], bf, acc[0][g]);
                acc[1][g] = wmma_f16(af[1], bf, acc[1][g]);
            }
        }
        __syncthreads();   // all waves done reading sH
        // elementwise gate math directly on D fragments
        int j = jt * 16 + ln;
#pragma unroll
        for (int mi = 0; mi < 2; ++mi) {
            int mt = mhalf * 2 + mi;
#pragma unroll
            for (int v = 0; v < 8; ++v) {
                int b = mt * 16 + hi * 8 + v;
                float ii = acc[mi][0][v], ff = acc[mi][1][v];
                float gg = acc[mi][2][v], oo = acc[mi][3][v];
                float c2 = sigf(ff) * cReg[mi][v] + sigf(ii) * tanhf(gg);
                float h2 = sigf(oo) * tanhf(c2);
                float mv = mask[t * 64 + b];
                float hold = (float)sH[b * 256 + j];
                float hn = (mv > 0.f) ? h2 : hold;
                cReg[mi][v] = (mv > 0.f) ? c2 : cReg[mi][v];
                sH[b * 256 + j] = (half_t)hn;
                y[((size_t)t * 64 + b) * HIDh + colOff + j] =
                    (half_t)((mv > 0.f) ? h2 : 0.f);
            }
        }
        __syncthreads();   // writes visible before next step's A-frag reads
    }
}

// ---------------------------------------------------------------------------
// LayerNorm over HID=512 + tag projection (K=32). One wave per (t,b) row;
// lane k computes feats[b,t,k]. W_tag transposed in LDS (conflict-free).
// ---------------------------------------------------------------------------
__global__ __launch_bounds__(256) void k_ln_tag(const half_t* __restrict__ y1,
                                                const float* __restrict__ ln_g,
                                                const float* __restrict__ ln_b,
                                                const float* __restrict__ W_tag,
                                                const float* __restrict__ b_tag,
                                                float* __restrict__ feats) {
    __shared__ float sW[512 * 32];   // transposed [j][k]
    __shared__ float sY[8][512];
    int tid = threadIdx.x;
    for (int idx = tid; idx < Kk * HIDh; idx += 256) {
        int k = idx >> 9, j = idx & 511;
        sW[j * 32 + k] = W_tag[idx];
    }
    __syncthreads();
    int wave = tid >> 5, lane = tid & 31;
    int r = blockIdx.x * 8 + wave;   // t*64 + b
    int t = r >> 6, b = r & 63;

    float s = 0.f, ss = 0.f;
    float xv[16];
#pragma unroll
    for (int i = 0; i < 16; ++i) {
        float x = (float)y1[(size_t)r * HIDh + i * 32 + lane];
        xv[i] = x; s += x; ss += x * x;
    }
#pragma unroll
    for (int o = 16; o >= 1; o >>= 1) {
        s  += __shfl_xor(s,  o, 32);
        ss += __shfl_xor(ss, o, 32);
    }
    float mu   = s / (float)HIDh;
    float var  = ss / (float)HIDh - mu * mu;
    float rstd = rsqrtf(var + 1e-5f);
#pragma unroll
    for (int i = 0; i < 16; ++i) {
        int j = i * 32 + lane;
        sY[wave][j] = (xv[i] - mu) * rstd * ln_g[j] + ln_b[j];
    }
    __syncthreads();
    float acc = b_tag[lane];
    for (int j = 0; j < HIDh; ++j) acc += sY[wave][j] * sW[j * 32 + lane];
    feats[((size_t)b * Tt + t) * Kk + lane] = acc;
}

// ---------------------------------------------------------------------------
// CRF NLL: one wave32 per batch row; lane = tag (K == 32 == wave width).
// ---------------------------------------------------------------------------
__global__ void k_crf(const float* __restrict__ feats, const int* __restrict__ tags,
                      const int* __restrict__ seq_len, const float* __restrict__ trans,
                      float* __restrict__ part) {
    __shared__ float sTr[32 * 32];
    int b = blockIdx.x;
    int lane = threadIdx.x;   // 32 threads
    for (int i = lane; i < 1024; i += 32) sTr[i] = trans[i];
    __syncthreads();

    int len = seq_len[b];
    float alpha = (lane == 30) ? 0.f : NEGV;    // START = 30
    for (int t = 0; t < len; ++t) {
        float emit = feats[((size_t)b * Tt + t) * Kk + lane];
        float vv[32];
        float mx = -3.4e38f;
#pragma unroll
        for (int j = 0; j < 32; ++j) {
            float aj = __shfl(alpha, j, 32);
            float v = aj + sTr[lane * 32 + j];
            vv[j] = v;
            mx = fmaxf(mx, v);
        }
        float sum = 0.f;
#pragma unroll
        for (int j = 0; j < 32; ++j) sum += expf(vv[j] - mx);
        alpha = mx + logf(sum) + emit;
    }
    // fwd = logsumexp_k(alpha[k] + trans[STOP,k]), STOP = 31
    float v = alpha + sTr[31 * 32 + lane];
    float mx = v;
#pragma unroll
    for (int o = 16; o >= 1; o >>= 1) mx = fmaxf(mx, __shfl_xor(mx, o, 32));
    float e = expf(v - mx);
#pragma unroll
    for (int o = 16; o >= 1; o >>= 1) e += __shfl_xor(e, o, 32);
    float fwd = mx + logf(e);
    // gold score (lane-parallel over time)
    float gp = 0.f;
    for (int t = lane; t < len; t += 32) {
        int prev = (t == 0) ? 30 : tags[b * Tt + t - 1];
        int cur  = tags[b * Tt + t];
        gp += sTr[cur * 32 + prev] + feats[((size_t)b * Tt + t) * Kk + cur];
    }
#pragma unroll
    for (int o = 16; o >= 1; o >>= 1) gp += __shfl_xor(gp, o, 32);
    if (lane == 0) {
        int last = (len > 0) ? tags[b * Tt + len - 1] : 30;
        float gold = gp + sTr[31 * 32 + last];
        part[b] = fwd - gold;
    }
}

__global__ void k_reduce(const float* __restrict__ part, float* __restrict__ out) {
    if (threadIdx.x == 0) {
        float s = 0.f;
        for (int i = 0; i < Bb; ++i) s += part[i];
        out[0] = s / (float)Bb;
    }
}

// ---------------------------------------------------------------------------
extern "C" void kernel_launch(void* const* d_in, const int* in_sizes, int n_in,
                              void* d_out, int out_size, void* d_ws, size_t ws_size,
                              hipStream_t stream) {
    (void)in_sizes; (void)n_in; (void)out_size; (void)ws_size;
    const int*   sentence = (const int*)d_in[0];
    const int*   seq_len  = (const int*)d_in[1];
    const int*   tags     = (const int*)d_in[2];
    const float* emb      = (const float*)d_in[3];
    const float* Wih0f = (const float*)d_in[4];
    const float* Whh0f = (const float*)d_in[5];
    const float* b0f   = (const float*)d_in[6];
    const float* Wih0b = (const float*)d_in[7];
    const float* Whh0b = (const float*)d_in[8];
    const float* b0b   = (const float*)d_in[9];
    const float* Wih1f = (const float*)d_in[10];
    const float* Whh1f = (const float*)d_in[11];
    const float* b1f   = (const float*)d_in[12];
    const float* Wih1b = (const float*)d_in[13];
    const float* Whh1b = (const float*)d_in[14];
    const float* b1b   = (const float*)d_in[15];
    const float* ln_g  = (const float*)d_in[16];
    const float* ln_b  = (const float*)d_in[17];
    const float* W_tag = (const float*)d_in[18];
    const float* b_tag = (const float*)d_in[19];
    const float* trans = (const float*)d_in[20];

    char* ws = (char*)d_ws;
    size_t off = 0;
    auto carve = [&](size_t bytes) { char* p = ws + off; off += (bytes + 255) & ~(size_t)255; return p; };

    half_t* x0   = (half_t*)carve((size_t)Tt * Bb * Ee * 2);
    half_t* y0   = (half_t*)carve((size_t)Tt * Bb * HIDh * 2);
    half_t* y1   = (half_t*)carve((size_t)Tt * Bb * HIDh * 2);
    float*  xgF  = (float*)carve((size_t)Tt * Bb * G4H * 4);
    float*  xgB  = (float*)carve((size_t)Tt * Bb * G4H * 4);
    half_t* pWih0f = (half_t*)carve((size_t)G4H * Ee * 2);
    half_t* pWih0b = (half_t*)carve((size_t)G4H * Ee * 2);
    half_t* pWhh0f = (half_t*)carve((size_t)G4H * Hh * 2);
    half_t* pWhh0b = (half_t*)carve((size_t)G4H * Hh * 2);
    half_t* pWih1f = (half_t*)carve((size_t)G4H * HIDh * 2);
    half_t* pWih1b = (half_t*)carve((size_t)G4H * HIDh * 2);
    half_t* pWhh1f = (half_t*)carve((size_t)G4H * Hh * 2);
    half_t* pWhh1b = (half_t*)carve((size_t)G4H * Hh * 2);
    float*  maskp  = (float*)carve((size_t)Tt * Bb * 4);
    float*  featsp = (float*)carve((size_t)Bb * Tt * Kk * 4);
    float*  partp  = (float*)carve((size_t)Bb * 4);

    // pack weights into WMMA B-fragment layout (f32 -> f16)
    k_pack_w<<<dim3(512), 256, 0, stream>>>(Wih0f, pWih0f, G4H, Ee);
    k_pack_w<<<dim3(512), 256, 0, stream>>>(Wih0b, pWih0b, G4H, Ee);
    k_pack_w<<<dim3(512), 256, 0, stream>>>(Whh0f, pWhh0f, G4H, Hh);
    k_pack_w<<<dim3(512), 256, 0, stream>>>(Whh0b, pWhh0b, G4H, Hh);
    k_pack_w<<<dim3(512), 256, 0, stream>>>(Wih1f, pWih1f, G4H, HIDh);
    k_pack_w<<<dim3(512), 256, 0, stream>>>(Wih1b, pWih1b, G4H, HIDh);
    k_pack_w<<<dim3(512), 256, 0, stream>>>(Whh1f, pWhh1f, G4H, Hh);
    k_pack_w<<<dim3(512), 256, 0, stream>>>(Whh1b, pWhh1b, G4H, Hh);

    k_embed<<<dim3(Tt * Bb), 256, 0, stream>>>(sentence, emb, x0);
    k_mask<<<dim3((Tt * Bb) / 256), 256, 0, stream>>>(seq_len, maskp);

    // layer 0: input-gate GEMMs (parallel over all T*B rows), then the scan
    k_gemm<<<dim3(256, 8), 256, 0, stream>>>(x0, pWih0f, b0f, xgF, Ee);
    k_gemm<<<dim3(256, 8), 256, 0, stream>>>(x0, pWih0b, b0b, xgB, Ee);
    k_scan<<<dim3(2), 1024, 0, stream>>>(xgF, xgB, pWhh0f, pWhh0b, maskp, y0);

    // layer 1
    k_gemm<<<dim3(256, 8), 256, 0, stream>>>(y0, pWih1f, b1f, xgF, HIDh);
    k_gemm<<<dim3(256, 8), 256, 0, stream>>>(y0, pWih1b, b1b, xgB, HIDh);
    k_scan<<<dim3(2), 1024, 0, stream>>>(xgF, xgB, pWhh1f, pWhh1b, maskp, y1);

    // layernorm + tag projection
    k_ln_tag<<<dim3((Tt * Bb) / 8), 256, 0, stream>>>(y1, ln_g, ln_b, W_tag, b_tag, featsp);

    // CRF NLL per batch row, then mean
    k_crf<<<dim3(Bb), 32, 0, stream>>>(featsp, tags, seq_len, trans, partp);
    k_reduce<<<dim3(1), 32, 0, stream>>>(partp, (float*)d_out);
}
